// Diffusion_29789893165499
// MI455X (gfx1250) — matrix operand order repaired and checked by reading, not compiled, not run on hardware
//
#include <hip/hip_runtime.h>
#include <math.h>

// ---------------------------------------------------------------------------
// Fused electron-nucleus message passing block for MI455X (gfx1250, wave32).
//
// Math restructure: agg[i,d] = sum_{k,j} edge_emb[i*256+k, j] * W_edge[j,d] * inp_spin[k,d]
//   =>  AGG = A @ B_spin   with A = edge_emb viewed as [1024 x 8192] (its natural layout)
//       and B_spin[k*32+j, d] = W_edge[j,d] * inp_spin[k,d] * scale1   (precomputed, f16)
// This turns the gather/elementwise/segment-sum into one dense WMMA GEMM (K=8192).
//
// Bandwidth-bound: ~37 MB unique reads vs 23.3 TB/s => ~1.6 us floor; all tiles
// are L2-resident (192 MB), so modest tile re-reads are cheap.
// ---------------------------------------------------------------------------

typedef _Float16 v16h __attribute__((ext_vector_type(16)));
typedef _Float16 v8h  __attribute__((ext_vector_type(8)));
typedef _Float16 v4h  __attribute__((ext_vector_type(4)));
typedef float    v8f  __attribute__((ext_vector_type(8)));
typedef float    v4f  __attribute__((ext_vector_type(4)));
typedef int      v4i  __attribute__((ext_vector_type(4)));

// Pointer types for the gfx1250 async-to-LDS DMA builtin (probe-discovered
// signature: (v4i addrspace(1)*, v4i addrspace(3)*, imm offset, imm cpol)).
typedef __attribute__((address_space(1))) v4i* gv4i_p;
typedef __attribute__((address_space(3))) v4i* lv4i_p;

#define N_EL   1024
#define N_NUC  256
#define DIM    256
#define EDIM   32
#define KDIM   (N_NUC * EDIM)   // 8192
#define KSTEPS (KDIM / 32)      // 256

// LDS row padding: 40 halves = 80 B stride -> bank stride 20; 20*r mod 64 is
// distinct for r = 0..15 => conflict-free 16-row fragment reads.
#define APAD 40
// Epilogue: 264 halves = 528 B stride -> bank stride 4, conflict-free.
#define EPAD 264

// SiLU gain from the reference's frozen init-time estimate (1/std(silu(N(0,1)))).
#define GAIN        1.7868f
#define INV_SQRT2F  0.70710678f

#if defined(__has_builtin)
#if __has_builtin(__builtin_amdgcn_global_load_async_to_lds_b128)
#define HAVE_ASYNC_LDS 1
#endif
#endif

__device__ __forceinline__ float silu_gain(float x) {
    float s = 1.0f / (1.0f + __expf(-x));
    return x * s * GAIN;
}

__device__ __forceinline__ v16h make_frag(v8h lo, v8h hi) {
    v16h r;
#pragma unroll
    for (int e = 0; e < 8; ++e) { r[e] = lo[e]; r[e + 8] = hi[e]; }
    return r;
}

__device__ __forceinline__ void wait_async0() {
#if defined(__has_builtin)
#if __has_builtin(__builtin_amdgcn_s_wait_asynccnt)
    __builtin_amdgcn_s_wait_asynccnt(0);
#else
    asm volatile("s_wait_asynccnt 0x0" ::: "memory");
#endif
#endif
}

// ---------------------------------------------------------------------------
// Kernel 1: precompute f16 B tables.
//  blocks [0,256):   Bt_up/Bt_dn row d:  Bt[d][k*32+j] = W_edge[j,d]*inp[k,d]*scale1
//  blocks [256,512): WtO/WtO2 row d:     Wt[d][k]      = W[k,d]   (transposed f16)
// ---------------------------------------------------------------------------
__global__ __launch_bounds__(256) void prep_kernel(
    const float* __restrict__ Wedge, const float* __restrict__ up,
    const float* __restrict__ dn,    const float* __restrict__ scale1,
    const float* __restrict__ Wout,  const float* __restrict__ Wout2,
    _Float16* __restrict__ BtUp, _Float16* __restrict__ BtDn,
    _Float16* __restrict__ WtO,  _Float16* __restrict__ WtO2)
{
    const int b = blockIdx.x, t = threadIdx.x;
    if (b < 256) {
        const int d  = b;
        const float s1 = scale1[0];
        const int j  = t & 31;      // edim index
        const int kk = t >> 5;      // nucleus sub-index (0..7)
        const float wj = Wedge[j * DIM + d] * s1;
#pragma unroll 4
        for (int kb = 0; kb < 32; ++kb) {
            const int k = kb * 8 + kk;
            const float u = up[(size_t)k * DIM + d];
            const float v = dn[(size_t)k * DIM + d];
            BtUp[(size_t)d * KDIM + k * EDIM + j] = (_Float16)(wj * u);
            BtDn[(size_t)d * KDIM + k * EDIM + j] = (_Float16)(wj * v);
        }
    } else {
        const int d = b - 256;      // output dim
        WtO [(size_t)d * DIM + t] = (_Float16)Wout [(size_t)t * DIM + d];
        WtO2[(size_t)d * DIM + t] = (_Float16)Wout2[(size_t)t * DIM + d];
    }
}

// ---------------------------------------------------------------------------
// Kernel 2: AGG[1024x256] = f16(edge_emb)[1024x8192] @ Bt_spin^T
// BM=64, BN=64, BK=32. 64 workgroups x 8 waves. Double-buffered LDS, one
// barrier per k-step; B tile staged via GLOBAL_LOAD_ASYNC_TO_LDS_B128
// (ASYNCcnt-tracked DMA), A tile via VGPR f32->f16 convert.
// ---------------------------------------------------------------------------
__global__ __launch_bounds__(256) void gemm_agg_kernel(
    const float* __restrict__ edge,        // flat [1024*8192] f32
    const _Float16* __restrict__ BtUp,     // [256][8192] f16 (d-major, K contiguous)
    const _Float16* __restrict__ BtDn,
    const int* __restrict__ nupPtr,
    float* __restrict__ agg)               // [1024][256] f32
{
    __shared__ _Float16 As[2][64][APAD];
    __shared__ _Float16 Bs[2][64][APAD];

    const int mblk = blockIdx.x & 15;
    const int nblk = blockIdx.x >> 4;
    const int m0 = mblk * 64, n0 = nblk * 64;
    const int nup = *nupPtr;
    const _Float16* __restrict__ Bt = (m0 < nup) ? BtUp : BtDn;

    const int t = threadIdx.x;
    const int lane = t & 31, w = t >> 5;
    const int mt = w & 3, nh = w >> 2;
    const int l16 = lane & 15, half = lane >> 4;

    // A-tile load indices (f32 -> f16): 512 float4 chunks, 2 per thread
    const int arow0 = t >> 3,         ac0 = (t & 7) << 2;
    const int arow1 = (t + 256) >> 3, ac1 = ((t + 256) & 7) << 2;
    // B-tile load indices: 256 chunks of 8 halves, 1 per thread
    const int brow = t >> 2, bch = (t & 3) << 3;

    const float*    __restrict__ aptr0 = &edge[(size_t)(m0 + arow0) * KDIM + ac0];
    const float*    __restrict__ aptr1 = &edge[(size_t)(m0 + arow1) * KDIM + ac1];
    const _Float16* __restrict__ bptr  = &Bt[(size_t)(n0 + brow) * KDIM + bch];

    const int frow = mt * 16 + l16;          // A fragment row in LDS
    const int b0r = (nh * 2 + 0) * 16 + l16; // B fragment rows in LDS
    const int b1r = (nh * 2 + 1) * 16 + l16;
    const int bko = 8 * half;                // B fragment K offset (same layout as A)

    v8f acc0 = {}; v8f acc1 = {};

#ifdef HAVE_ASYNC_LDS
#define STAGE_B(K0, BUF)                                                              \
    __builtin_amdgcn_global_load_async_to_lds_b128(                                   \
        (gv4i_p)(bptr + (K0)),                                                        \
        (lv4i_p)&Bs[BUF][brow][bch], 0, 0)
#else
#define STAGE_B(K0, BUF)                                                              \
    do {                                                                              \
        v8h b8_ = *reinterpret_cast<const v8h*>(bptr + (K0));                         \
        *reinterpret_cast<v8h*>(&Bs[BUF][brow][bch]) = b8_;                           \
    } while (0)
#endif

#define STAGE_TILES(K0, BUF)                                                          \
    do {                                                                              \
        v4f a4_ = *reinterpret_cast<const v4f*>(aptr0 + (K0));                        \
        v4h h_; h_[0] = (_Float16)a4_[0]; h_[1] = (_Float16)a4_[1];                   \
                h_[2] = (_Float16)a4_[2]; h_[3] = (_Float16)a4_[3];                   \
        *reinterpret_cast<v4h*>(&As[BUF][arow0][ac0]) = h_;                           \
        v4f c4_ = *reinterpret_cast<const v4f*>(aptr1 + (K0));                        \
        v4h g_; g_[0] = (_Float16)c4_[0]; g_[1] = (_Float16)c4_[1];                   \
                g_[2] = (_Float16)c4_[2]; g_[3] = (_Float16)c4_[3];                   \
        *reinterpret_cast<v4h*>(&As[BUF][arow1][ac1]) = g_;                           \
        STAGE_B(K0, BUF);                                                             \
    } while (0)

    // Prologue: stage k-step 0 into buffer 0.
    STAGE_TILES(0, 0);

    for (int ks = 0; ks < KSTEPS; ++ks) {
        const int cur = ks & 1;
#ifdef HAVE_ASYNC_LDS
        wait_async0();              // our async B-tile DMA for buffer `cur` landed
#endif
        __syncthreads();            // all waves' stores to buffer `cur` visible
        if (ks + 1 < KSTEPS) {
            STAGE_TILES((ks + 1) * 32, cur ^ 1);   // overlaps with WMMAs below
            __builtin_prefetch(aptr0 + (ks + 2) * 32, 0, 0);  // global_prefetch_b8
        }
        // Fragments per ISA 16-bit layouts (A and B: K = 8*half+e / 16+8*half+e)
        v8h alo = *reinterpret_cast<const v8h*>(&As[cur][frow][8 * half]);
        v8h ahi = *reinterpret_cast<const v8h*>(&As[cur][frow][16 + 8 * half]);
        v16h a = make_frag(alo, ahi);
        v16h b0 = make_frag(*reinterpret_cast<const v8h*>(&Bs[cur][b0r][bko]),
                            *reinterpret_cast<const v8h*>(&Bs[cur][b0r][bko + 16]));
        v16h b1 = make_frag(*reinterpret_cast<const v8h*>(&Bs[cur][b1r][bko]),
                            *reinterpret_cast<const v8h*>(&Bs[cur][b1r][bko + 16]));
        acc0 = __builtin_amdgcn_wmma_f32_16x16x32_f16(false, a, false, b0, (short)0, acc0, false, false);
        acc1 = __builtin_amdgcn_wmma_f32_16x16x32_f16(false, a, false, b1, (short)0, acc1, false, false);
    }

    // C layout: lane holds col N = l16, rows M = 8*half + r
#pragma unroll
    for (int r = 0; r < 8; ++r) {
        const int m = m0 + mt * 16 + half * 8 + r;
        agg[(size_t)m * DIM + n0 + (nh * 2 + 0) * 16 + l16] = acc0[r];
        agg[(size_t)m * DIM + n0 + (nh * 2 + 1) * 16 + l16] = acc1[r];
    }
#undef STAGE_TILES
#undef STAGE_B
}

// ---------------------------------------------------------------------------
// Kernel 3: epilogue. Per 16 electrons:
//   h  = act( (elec@W_out + b_out + agg*norm) * scale2 )
//   h2 = act( h@W_out2 + b_out2 )
//   out = (elec + h2) * inv_sqrt2
// 64 WGs x 8 waves; each wave owns 2 n-tiles (16x16), K=256.
// ---------------------------------------------------------------------------
__global__ __launch_bounds__(256) void epilogue_kernel(
    const float* __restrict__ elec, const float* __restrict__ agg,
    const float* __restrict__ norm,
    const _Float16* __restrict__ WtO,  const float* __restrict__ bO,
    const _Float16* __restrict__ WtO2, const float* __restrict__ bO2,
    const float* __restrict__ scale2p, float* __restrict__ out)
{
    __shared__ _Float16 Ah[16][EPAD];
    __shared__ _Float16 Hh[16][EPAD];

    const int r0 = blockIdx.x << 4;
    const int t = threadIdx.x, lane = t & 31, w = t >> 5;
    const int l16 = lane & 15, half = lane >> 4;
    const float s2 = scale2p[0];

    // Stage f16(elec_emb) rows
#pragma unroll
    for (int i = 0; i < 2; ++i) {
        const int f = t + i * 256;           // float4 index in [0,512)
        const int row = f >> 6, c4 = (f & 63) << 2;
        v4f a4 = *reinterpret_cast<const v4f*>(&elec[(size_t)(r0 + row) * DIM + c4]);
        v4h h; h[0] = (_Float16)a4[0]; h[1] = (_Float16)a4[1];
               h[2] = (_Float16)a4[2]; h[3] = (_Float16)a4[3];
        *reinterpret_cast<v4h*>(&Ah[row][c4]) = h;
    }
    __syncthreads();

    const int n0c = (w * 2 + 0) * 16 + l16;
    const int n1c = (w * 2 + 1) * 16 + l16;

    v8f acc0 = {}, acc1 = {};
#pragma unroll
    for (int ks = 0; ks < 8; ++ks) {
        const int k0 = ks * 32;
        v16h a = make_frag(*reinterpret_cast<const v8h*>(&Ah[l16][k0 + 8 * half]),
                           *reinterpret_cast<const v8h*>(&Ah[l16][k0 + 16 + 8 * half]));
        const _Float16* p0 = &WtO[(size_t)n0c * DIM + k0 + 8 * half];
        const _Float16* p1 = &WtO[(size_t)n1c * DIM + k0 + 8 * half];
        v16h b0 = make_frag(*reinterpret_cast<const v8h*>(p0), *reinterpret_cast<const v8h*>(p0 + 16));
        v16h b1 = make_frag(*reinterpret_cast<const v8h*>(p1), *reinterpret_cast<const v8h*>(p1 + 16));
        acc0 = __builtin_amdgcn_wmma_f32_16x16x32_f16(false, a, false, b0, (short)0, acc0, false, false);
        acc1 = __builtin_amdgcn_wmma_f32_16x16x32_f16(false, a, false, b1, (short)0, acc1, false, false);
    }

    {   // bias + agg*norm, scale2, activation -> Hh (f16)
        const float bias0 = bO[n0c], bias1 = bO[n1c];
#pragma unroll
        for (int r = 0; r < 8; ++r) {
            const int m = half * 8 + r, gm = r0 + m;
            const float nm = norm[gm];
            float v0 = (acc0[r] + bias0 + agg[(size_t)gm * DIM + n0c] * nm) * s2;
            float v1 = (acc1[r] + bias1 + agg[(size_t)gm * DIM + n1c] * nm) * s2;
            Hh[m][n0c] = (_Float16)silu_gain(v0);
            Hh[m][n1c] = (_Float16)silu_gain(v1);
        }
    }
    __syncthreads();

    v8f d0 = {}, d1 = {};
#pragma unroll
    for (int ks = 0; ks < 8; ++ks) {
        const int k0 = ks * 32;
        v16h a = make_frag(*reinterpret_cast<const v8h*>(&Hh[l16][k0 + 8 * half]),
                           *reinterpret_cast<const v8h*>(&Hh[l16][k0 + 16 + 8 * half]));
        const _Float16* p0 = &WtO2[(size_t)n0c * DIM + k0 + 8 * half];
        const _Float16* p1 = &WtO2[(size_t)n1c * DIM + k0 + 8 * half];
        v16h b0 = make_frag(*reinterpret_cast<const v8h*>(p0), *reinterpret_cast<const v8h*>(p0 + 16));
        v16h b1 = make_frag(*reinterpret_cast<const v8h*>(p1), *reinterpret_cast<const v8h*>(p1 + 16));
        d0 = __builtin_amdgcn_wmma_f32_16x16x32_f16(false, a, false, b0, (short)0, d0, false, false);
        d1 = __builtin_amdgcn_wmma_f32_16x16x32_f16(false, a, false, b1, (short)0, d1, false, false);
    }
    {   // bias2, activation, residual, write out
        const float bias0 = bO2[n0c], bias1 = bO2[n1c];
#pragma unroll
        for (int r = 0; r < 8; ++r) {
            const int m = half * 8 + r, gm = r0 + m;
            const float v0 = silu_gain(d0[r] + bias0);
            const float v1 = silu_gain(d1[r] + bias1);
            out[(size_t)gm * DIM + n0c] = (elec[(size_t)gm * DIM + n0c] + v0) * INV_SQRT2F;
            out[(size_t)gm * DIM + n1c] = (elec[(size_t)gm * DIM + n1c] + v1) * INV_SQRT2F;
        }
    }
}

// ---------------------------------------------------------------------------
// Launcher. Workspace layout (bytes):
//   [0,        4M)  Bt_up   f16 [256][8192]
//   [4M,       8M)  Bt_dn   f16 [256][8192]
//   [8M,  8M+128K)  WtO     f16 [256][256]
//   [.. , +128K  )  WtO2    f16 [256][256]
//   [.. , +1M    )  AGG     f32 [1024][256]
// Total ~9.25 MB.
// ---------------------------------------------------------------------------
extern "C" void kernel_launch(void* const* d_in, const int* in_sizes, int n_in,
                              void* d_out, int out_size, void* d_ws, size_t ws_size,
                              hipStream_t stream) {
    const float* elec   = (const float*)d_in[0];
    const float* up     = (const float*)d_in[1];
    const float* dn     = (const float*)d_in[2];
    const float* edge   = (const float*)d_in[3];
    // d_in[4] = contr: runtime-dead in the reference (init-time only)
    const float* norm   = (const float*)d_in[5];
    const float* Wout   = (const float*)d_in[6];
    const float* bout   = (const float*)d_in[7];
    const float* Wedge  = (const float*)d_in[8];
    const float* Wout2  = (const float*)d_in[9];
    const float* bout2  = (const float*)d_in[10];
    const float* scale1 = (const float*)d_in[11];
    const float* scale2 = (const float*)d_in[12];
    const int*   nup    = (const int*)d_in[13];

    char* ws = (char*)d_ws;
    const size_t BT_BYTES = (size_t)DIM * KDIM * sizeof(_Float16);   // 4 MB
    _Float16* BtUp = (_Float16*)(ws);
    _Float16* BtDn = (_Float16*)(ws + BT_BYTES);
    _Float16* WtO  = (_Float16*)(ws + 2 * BT_BYTES);
    _Float16* WtO2 = (_Float16*)(ws + 2 * BT_BYTES + (size_t)DIM * DIM * 2);
    float*    aggw = (float*)   (ws + 2 * BT_BYTES + (size_t)DIM * DIM * 4);

    prep_kernel<<<512, 256, 0, stream>>>(Wedge, up, dn, scale1, Wout, Wout2,
                                         BtUp, BtDn, WtO, WtO2);
    gemm_agg_kernel<<<64, 256, 0, stream>>>(edge, BtUp, BtDn, nup, aggw);
    epilogue_kernel<<<64, 256, 0, stream>>>(elec, aggw, norm, WtO, bout,
                                            WtO2, bout2, scale2, (float*)d_out);
}
